// Attention_85315230368481
// MI455X (gfx1250) — matrix operand order repaired and checked by reading, not compiled, not run on hardware
//
#include <hip/hip_runtime.h>
#include <hip/hip_bf16.h>
#include <stdint.h>

// ---------------- problem constants ----------------
constexpr int Bc  = 2;
constexpr int Sc  = 2048;
constexpr int Dc  = 2048;
constexpr int Hc  = 16;
constexpr int HDc = 128;           // head dim
constexpr int MROWS = Bc * Sc;     // 4096 rows for the [B*S, D] GEMMs

typedef __bf16  bf16;
typedef __bf16  v16bf __attribute__((ext_vector_type(16)));
typedef float   v8f   __attribute__((ext_vector_type(8)));

union FragB { uint32_t u[8]; v16bf v; };  // 16 bf16 operand fragment
union FragF { float    f[8]; v8f  v; };   // 16x16 f32 accumulator fragment

// ---------------- fragment loaders (per CDNA5 ISA §7.12.2 layouts) ----------------
// A matrix 16x32 bf16, row-major source with leading dim ld (elements).
// lane L (<16): row M=L, K = {2v} v=0..3 and {16+2v}; lane L+16: same row, K+8.
__device__ inline void load_a_frag(FragB& f, const bf16* base, int ld, int lane) {
    const int l15 = lane & 15, lh = (lane >> 4) & 1;
    const bf16* rowp = base + (size_t)l15 * ld;
    const int kb = lh * 8;
#pragma unroll
    for (int v = 0; v < 4; ++v)
        f.u[v] = *(const uint32_t*)(rowp + kb + 2 * v);
#pragma unroll
    for (int v = 0; v < 4; ++v)
        f.u[4 + v] = *(const uint32_t*)(rowp + 16 + kb + 2 * v);
}

// B matrix 32x16 bf16 fetched from a K-contiguous (transposed, [N][K]) source.
// lane n (<16): column N=n, K = {2v} v=0..3 and {16+2v}; lane n+16: same column, K+8.
__device__ inline void load_bt_frag(FragB& f, const bf16* base, int ld, int lane) {
    const int l15 = lane & 15, lh = (lane >> 4) & 1;
    const bf16* rowp = base + (size_t)l15 * ld + lh * 8;
#pragma unroll
    for (int v = 0; v < 4; ++v)
        f.u[v] = *(const uint32_t*)(rowp + 2 * v);
#pragma unroll
    for (int v = 0; v < 4; ++v)
        f.u[4 + v] = *(const uint32_t*)(rowp + 16 + 2 * v);
}

__device__ inline v8f wmma_bf16(const FragB& a, const FragB& b, const v8f& c) {
    return __builtin_amdgcn_wmma_f32_16x16x32_bf16(false, a.v, false, b.v,
                                                   (short)0, c, false, false);
}

// CDNA5 async global->LDS copy, 16 bytes per lane (ASYNCcnt-tracked).
// vdst = VGPR with LDS byte address, vaddr = 64-bit global address, GV mode.
__device__ inline void async_copy_b128(uint32_t lds_addr, const void* gptr) {
    asm volatile("global_load_async_to_lds_b128 %0, %1, off"
                 :: "v"(lds_addr), "v"(gptr) : "memory");
}
__device__ inline void wait_asynccnt0() {
    asm volatile("s_wait_asynccnt 0x0" ::: "memory");
}

// ---------------- elementwise fp32 -> bf16 ----------------
__global__ __launch_bounds__(256) void cvt_kernel(const float* __restrict__ in,
                                                  bf16* __restrict__ out, int n) {
    int i = blockIdx.x * 256 + threadIdx.x;
    if (i < n) out[i] = (bf16)in[i];
}

// ---------------- weight transpose + convert: Wt[n][k] = (bf16) W[k][n], 2048x2048 ----------------
__global__ __launch_bounds__(256) void wtrans_kernel(const float* __restrict__ W,
                                                     bf16* __restrict__ Wt) {
    __shared__ float tile[32][33];
    const int tx = threadIdx.x, ty = threadIdx.y;
    const int gx = blockIdx.x * 32;  // n
    const int gy = blockIdx.y * 32;  // k
#pragma unroll
    for (int i = 0; i < 4; ++i)
        tile[ty + 8 * i][tx] = W[(size_t)(gy + ty + 8 * i) * Dc + gx + tx];
    __syncthreads();
#pragma unroll
    for (int i = 0; i < 4; ++i)
        Wt[(size_t)(gx + ty + 8 * i) * Dc + gy + tx] = (bf16)tile[tx][ty + 8 * i];
}

// ---------------- bf16 GEMM:  C[M,N](f32) = A[M,K](bf16) * BT[N,K](bf16)^T ----------------
// Block: 256 thr = 8 waves (4x2); wave tile 32x32; block tile 128x64.
// A/B block tiles staged in LDS via async global->LDS copies, double buffered.
__global__ __launch_bounds__(256) void gemm_kernel(const bf16* __restrict__ A,
                                                   const bf16* __restrict__ BT,
                                                   float* __restrict__ C,
                                                   int Ndim, int Kdim) {
    __shared__ bf16 As[2][128][32];   // [buf][m][k]  8KB per buf
    __shared__ bf16 Bs[2][64][32];    // [buf][n][k]  4KB per buf

    const int tid  = threadIdx.x;
    const int wid  = tid >> 5;
    const int lane = tid & 31;
    const int wm = wid >> 1, wn = wid & 1;
    const size_t rowBlk = (size_t)blockIdx.x * 128;
    const size_t colBlk = (size_t)blockIdx.y * 64;

    const uint32_t asBase = (uint32_t)(uintptr_t)(&As[0][0][0]);
    const uint32_t bsBase = (uint32_t)(uintptr_t)(&Bs[0][0][0]);

    // issue the async copies for k-slice k0 into buffer `buf` (3 x b128 per thread)
    auto issue = [&](int buf, int k0) {
#pragma unroll
        for (int i = 0; i < 2; ++i) {                       // A tile: 512 segs of 16B
            int s   = tid + i * 256;
            int row = s >> 2, c8 = (s & 3) * 8;
            const bf16* g = A + (rowBlk + row) * Kdim + k0 + c8;
            async_copy_b128(asBase + (uint32_t)buf * (128 * 32 * 2) + (uint32_t)s * 16, g);
        }
        {                                                   // B tile: 256 segs of 16B
            int row = tid >> 2, c8 = (tid & 3) * 8;
            const bf16* g = BT + (colBlk + row) * Kdim + k0 + c8;
            async_copy_b128(bsBase + (uint32_t)buf * (64 * 32 * 2) + (uint32_t)tid * 16, g);
        }
    };

    FragF c[2][2];
#pragma unroll
    for (int i = 0; i < 2; ++i)
#pragma unroll
        for (int j = 0; j < 2; ++j)
#pragma unroll
            for (int r = 0; r < 8; ++r) c[i][j].f[r] = 0.f;

    const int nk = Kdim / 32;
    issue(0, 0);

    for (int it = 0; it < nk; ++it) {
        const int cur = it & 1;
        wait_asynccnt0();        // this wave's copies into `cur` are complete
        __syncthreads();         // everyone's copies complete; prior reads of `cur^1` done
        if (it + 1 < nk) issue(cur ^ 1, (it + 1) * 32);

        FragB a0, a1, b0, b1;
        load_a_frag (a0, &As[cur][wm * 32][0],      32, lane);
        load_a_frag (a1, &As[cur][wm * 32 + 16][0], 32, lane);
        load_bt_frag(b0, &Bs[cur][wn * 32][0],      32, lane);
        load_bt_frag(b1, &Bs[cur][wn * 32 + 16][0], 32, lane);
        c[0][0].v = wmma_bf16(a0, b0, c[0][0].v);
        c[0][1].v = wmma_bf16(a0, b1, c[0][1].v);
        c[1][0].v = wmma_bf16(a1, b0, c[1][0].v);
        c[1][1].v = wmma_bf16(a1, b1, c[1][1].v);
    }

    const int l15 = lane & 15, lh = (lane >> 4) & 1;
#pragma unroll
    for (int i = 0; i < 2; ++i)
#pragma unroll
        for (int j = 0; j < 2; ++j)
#pragma unroll
            for (int r = 0; r < 8; ++r) {
                size_t m = rowBlk + wm * 32 + i * 16 + r + 8 * lh;
                size_t n = colBlk + wn * 32 + j * 16 + l15;
                C[m * Ndim + n] = c[i][j].f[r];
            }
}

// ---------------- RoPE + [B,S,H,HD] -> bf16 [B,H,S,HD] ----------------
__global__ __launch_bounds__(256) void rope_kernel(const float* __restrict__ proj,
                                                   const float* __restrict__ cosT,
                                                   const float* __restrict__ sinT,
                                                   bf16* __restrict__ outR) {
    int idx = blockIdx.x * 256 + threadIdx.x;      // B*H*S*(HD/2) threads
    const int P = HDc / 2;
    int p = idx % P;
    int s = (idx / P) % Sc;
    int h = (idx / (P * Sc)) % Hc;
    int b =  idx / (P * Sc * Hc);
    const float* src = proj + ((size_t)(b * Sc + s)) * Dc + h * HDc + 2 * p;
    float te = src[0], to = src[1];
    float cc = cosT[s * P + p], ss = sinT[s * P + p];
    bf16* dst = outR + (((size_t)(b * Hc + h)) * Sc + s) * HDc + 2 * p;
    dst[0] = (bf16)(te * cc - to * ss);
    dst[1] = (bf16)(te * ss + to * cc);
}

// ---------------- V: [B,S,H,HD] f32 -> bf16 [B,H,HD,S] ----------------
__global__ __launch_bounds__(256) void vtrans_kernel(const float* __restrict__ proj,
                                                     bf16* __restrict__ Vt) {
    int idx = blockIdx.x * 256 + threadIdx.x;      // B*H*S*HD threads
    int hd = idx % HDc;
    int s  = (idx / HDc) % Sc;
    int h  = (idx / (HDc * Sc)) % Hc;
    int b  =  idx / (HDc * Sc * Hc);
    float v = proj[((size_t)(b * Sc + s)) * Dc + h * HDc + hd];
    Vt[(((size_t)(b * Hc + h)) * HDc + hd) * Sc + s] = (bf16)v;
}

// ---------------- Flash attention: Q,K [B,H,S,HD] bf16; Vt [B,H,HD,S] bf16 ----------------
// grid (S/64, H, B); block 128 (4 waves); wave = one 16-row query tile.
__global__ __launch_bounds__(128) void attn_kernel(const bf16* __restrict__ Q,
                                                   const bf16* __restrict__ K,
                                                   const bf16* __restrict__ Vt,
                                                   bf16* __restrict__ O) {
    __shared__ bf16 lds_p[4][16][32];
    const int wid  = threadIdx.x >> 5;
    const int lane = threadIdx.x & 31;
    const int l15 = lane & 15, lh = (lane >> 4) & 1;
    const int b = blockIdx.z, h = blockIdx.y;
    const int q0 = blockIdx.x * 64 + wid * 16;

    const size_t headOff = ((size_t)(b * Hc + h)) * Sc * HDc;
    const bf16* Qh = Q  + headOff;
    const bf16* Kh = K  + headOff;
    const bf16* Vh = Vt + headOff;                 // [HD][S]

    FragB qa[4];
#pragma unroll
    for (int st = 0; st < 4; ++st)
        load_a_frag(qa[st], Qh + (size_t)q0 * HDc + st * 32, HDc, lane);

    FragF o[8];
    float mrun[8], lrun[8];
#pragma unroll
    for (int r = 0; r < 8; ++r) {
        mrun[r] = -1e30f; lrun[r] = 0.f;
#pragma unroll
        for (int j = 0; j < 8; ++j) o[j].f[r] = 0.f;
    }

    const float SCALE = 0.08838834764831845f;      // 1/sqrt(128)
    const int nChunks = (q0 + 16 + 31) / 32;       // causal: keys 0 .. q0+15

    for (int c = 0; c < nChunks; ++c) {
        const int kb = c * 32;
        FragF s0, s1;
#pragma unroll
        for (int r = 0; r < 8; ++r) { s0.f[r] = 0.f; s1.f[r] = 0.f; }

        // scores = Q (16xHD) * K^T (HDx32), contracted in 4 steps of 32
#pragma unroll
        for (int st = 0; st < 4; ++st) {
            FragB kb0, kb1;
            load_bt_frag(kb0, Kh + (size_t)kb        * HDc + st * 32, HDc, lane);
            load_bt_frag(kb1, Kh + (size_t)(kb + 16) * HDc + st * 32, HDc, lane);
            s0.v = wmma_bf16(qa[st], kb0, s0.v);
            s1.v = wmma_bf16(qa[st], kb1, s1.v);
        }

        // online softmax over this 32-key chunk
#pragma unroll
        for (int r = 0; r < 8; ++r) {
            const int m  = r + 8 * lh;
            const int qg = q0 + m;
            const int kg0 = kb + l15, kg1 = kb + 16 + l15;
            float v0 = s0.f[r] * SCALE + (kg0 <= qg ? 0.f : -1e9f);
            float v1 = s1.f[r] * SCALE + (kg1 <= qg ? 0.f : -1e9f);
            float mx = fmaxf(v0, v1);
            mx = fmaxf(mx, __shfl_xor(mx, 1));
            mx = fmaxf(mx, __shfl_xor(mx, 2));
            mx = fmaxf(mx, __shfl_xor(mx, 4));
            mx = fmaxf(mx, __shfl_xor(mx, 8));
            float mnew = fmaxf(mrun[r], mx);
            float corr = __expf(mrun[r] - mnew);
            float e0 = __expf(v0 - mnew);
            float e1 = __expf(v1 - mnew);
            float rs = e0 + e1;
            rs += __shfl_xor(rs, 1);
            rs += __shfl_xor(rs, 2);
            rs += __shfl_xor(rs, 4);
            rs += __shfl_xor(rs, 8);
            lrun[r] = lrun[r] * corr + rs;
            mrun[r] = mnew;
#pragma unroll
            for (int j = 0; j < 8; ++j) o[j].f[r] *= corr;
            // stage P (C-layout) into LDS to regather as an A fragment
            lds_p[wid][m][l15]      = (bf16)e0;
            lds_p[wid][m][16 + l15] = (bf16)e1;
        }
        asm volatile("s_wait_dscnt 0" ::: "memory");

        FragB pa;
        load_a_frag(pa, &lds_p[wid][0][0], 32, lane);
        asm volatile("s_wait_dscnt 0" ::: "memory");

        // O += P (16x32) * V (32xHD): 8 N-tiles of 16
#pragma unroll
        for (int j = 0; j < 8; ++j) {
            FragB vb;
            load_bt_frag(vb, Vh + (size_t)(j * 16) * Sc + kb, Sc, lane);
            o[j].v = wmma_bf16(pa, vb, o[j].v);
        }
    }

    // normalize and write bf16 [B,S,H*HD]
#pragma unroll
    for (int r = 0; r < 8; ++r) {
        const int qg = q0 + r + 8 * lh;
        const float inv = 1.0f / lrun[r];
        bf16* orow = O + ((size_t)b * Sc + qg) * Dc + h * HDc;
#pragma unroll
        for (int j = 0; j < 8; ++j)
            orow[j * 16 + l15] = (bf16)(o[j].f[r] * inv);
    }
}

// ---------------- host launch ----------------
extern "C" void kernel_launch(void* const* d_in, const int* in_sizes, int n_in,
                              void* d_out, int out_size, void* d_ws, size_t ws_size,
                              hipStream_t stream) {
    const float* x    = (const float*)d_in[0];
    const float* Wq   = (const float*)d_in[1];
    const float* Wk   = (const float*)d_in[2];
    const float* Wv   = (const float*)d_in[3];
    const float* Wo   = (const float*)d_in[4];
    const float* fcos = (const float*)d_in[5];
    const float* fsin = (const float*)d_in[6];
    float* out = (float*)d_out;

    char* ws = (char*)d_ws;
    size_t off = 0;
    auto alloc = [&](size_t bytes) -> void* {
        void* p = ws + off;
        off += (bytes + 255) & ~(size_t)255;
        return p;
    };
    const size_t NX = (size_t)Bc * Sc * Dc;   // 8.4M elements
    const size_t NW = (size_t)Dc * Dc;        // 4.2M elements

    bf16*  xb    = (bf16*)alloc(NX * sizeof(bf16));      // bf16 x
    bf16*  Wt    = (bf16*)alloc(NW * sizeof(bf16));      // transposed weight (reused)
    float* proj  = (float*)alloc(NX * sizeof(float));    // GEMM f32 output (reused)
    bf16*  Qr    = (bf16*)alloc(NX * sizeof(bf16));      // [B,H,S,HD]
    bf16*  Kr    = (bf16*)alloc(NX * sizeof(bf16));      // [B,H,S,HD]
    bf16*  Vt    = (bf16*)alloc(NX * sizeof(bf16));      // [B,H,HD,S]
    bf16*  attnO = (bf16*)alloc(NX * sizeof(bf16));      // [B,S,D]

    const dim3 tb256(256), tb128(128);
    const dim3 gW(64, 64), bW(32, 8);
    const dim3 gGemm(MROWS / 128, Dc / 64);              // 32 x 32
    const int  gCvt   = (int)(NX / 256);
    const int  gRope  = Bc * Hc * Sc * (HDc / 2) / 256;
    const int  gVtr   = (int)(NX / 256);
    const dim3 gAttn(Sc / 64, Hc, Bc);

    // x -> bf16
    cvt_kernel<<<gCvt, tb256, 0, stream>>>(x, xb, (int)NX);

    // Q = rope(x @ Wq)
    wtrans_kernel<<<gW, bW, 0, stream>>>(Wq, Wt);
    gemm_kernel<<<gGemm, tb256, 0, stream>>>(xb, Wt, proj, Dc, Dc);
    rope_kernel<<<gRope, tb256, 0, stream>>>(proj, fcos, fsin, Qr);

    // K = rope(x @ Wk)
    wtrans_kernel<<<gW, bW, 0, stream>>>(Wk, Wt);
    gemm_kernel<<<gGemm, tb256, 0, stream>>>(xb, Wt, proj, Dc, Dc);
    rope_kernel<<<gRope, tb256, 0, stream>>>(proj, fcos, fsin, Kr);

    // V = (x @ Wv) transposed per head
    wtrans_kernel<<<gW, bW, 0, stream>>>(Wv, Wt);
    gemm_kernel<<<gGemm, tb256, 0, stream>>>(xb, Wt, proj, Dc, Dc);
    vtrans_kernel<<<gVtr, tb256, 0, stream>>>(proj, Vt);

    // flash attention
    attn_kernel<<<gAttn, tb128, 0, stream>>>(Qr, Kr, Vt, attnO);

    // out = attnO @ Wo  (f32 output)
    wtrans_kernel<<<gW, bW, 0, stream>>>(Wo, Wt);
    gemm_kernel<<<gGemm, tb256, 0, stream>>>(attnO, Wt, out, Dc, Dc);
}
